// MultiheadAttention_42073499632235
// MI455X (gfx1250) — compile-verified, matrix operand order checked by reading, not run on hardware
//
#include <hip/hip_runtime.h>
#include <hip/hip_bf16.h>

// ---------------------------------------------------------------------------
// MI455X (gfx1250) relative-position multi-head attention.
// B=4 S=1024 E=512 H=8 D=64 MAXP=512. wave32, WMMA bf16 16x16x32, f32 accum.
// Bandwidth-bound on the 134MB attn output -> single-touch attn, LDS-resident
// per-q-tile rel tables, bf16 WMMA everywhere, async-to-LDS V staging.
// ---------------------------------------------------------------------------

typedef __attribute__((ext_vector_type(16))) __bf16 v16bf;
typedef __attribute__((ext_vector_type(8)))  __bf16 v8bf;
typedef __attribute__((ext_vector_type(4)))  __bf16 v4bf;
typedef __attribute__((ext_vector_type(8)))  float  v8f;
typedef __attribute__((ext_vector_type(4)))  float  v4f;
typedef int v4i_ __attribute__((vector_size(16)));   // builtin's pointee type

#if defined(__AMDGCN__) && __has_builtin(__builtin_amdgcn_global_load_async_to_lds_b128)
#define ASYNC_CP 1
#else
#define ASYNC_CP 0
#endif

#if defined(__AMDGCN__) && __has_builtin(__builtin_amdgcn_s_wait_asynccnt)
#define WAIT_ASYNC(n) __builtin_amdgcn_s_wait_asynccnt(n)
#elif defined(__AMDGCN__)
#define WAIT_ASYNC(n) asm volatile("s_wait_asynccnt " #n ::: "memory")
#else
#define WAIT_ASYNC(n)
#endif

#if ASYNC_CP
// per-lane 16B async copy global -> LDS, tracked by ASYNCcnt.
// Signature (from clang diagnostic): (v4i addrspace(1)*, v4i addrspace(3)*,
// imm offset, imm cpol); HIP prints AS1 as __device__.
#define ASYNC_B128(gsrc, ldst)                                               \
  __builtin_amdgcn_global_load_async_to_lds_b128(                            \
      (__attribute__((address_space(1))) v4i_*)(void*)(gsrc),                \
      (__attribute__((address_space(3))) v4i_*)(void*)(ldst), 0, 0)
#endif

// D = A(16x32 bf16) * B(32x16 bf16) + C(16x16 f32)
__device__ __forceinline__ v8f wmma_bf16(v16bf a, v16bf b, v8f c) {
  return __builtin_amdgcn_wmma_f32_16x16x32_bf16(
      false, a, false, b, (short)0, c, false, false);
}

// Load a 16x32 bf16 tile (row-major, row stride `ld`, 16B-aligned rows) into
// the WMMA A/B per-lane fragment layout (ISA 7.12.2, 16-bit matrices):
// lanes 0-15 : row = lane,     K = {0..7, 16..23}
// lanes 16-31: row = lane-16,  K = {8..15, 24..31}
__device__ __forceinline__ v16bf load_frag(const __bf16* p, int ld) {
  const int lane = threadIdx.x & 31;
  const int row  = lane & 15;
  const int koff = (lane >> 4) << 3;
  const __bf16* r = p + row * ld + koff;
  v8bf c0 = *(const v8bf*)(r);
  v8bf c1 = *(const v8bf*)(r + 16);
  return __builtin_shufflevector(c0, c1, 0,1,2,3,4,5,6,7,8,9,10,11,12,13,14,15);
}

__device__ __forceinline__ v4bf cvt4(v4f x) {
  v4bf y;
  y[0] = (__bf16)x[0]; y[1] = (__bf16)x[1];
  y[2] = (__bf16)x[2]; y[3] = (__bf16)x[3];
  return y;
}
__device__ __forceinline__ v4f vmax4(v4f a, v4f b) {
  v4f r;
  r[0] = fmaxf(a[0], b[0]); r[1] = fmaxf(a[1], b[1]);
  r[2] = fmaxf(a[2], b[2]); r[3] = fmaxf(a[3], b[3]);
  return r;
}

// ---------------------------------------------------------------------------
// K0: tables -> bf16; Tkb [1056][64] padded; Tvt [64][1056] transposed+padded.
// ---------------------------------------------------------------------------
__global__ void prep_tables(const float* __restrict__ tab_k,
                            const float* __restrict__ tab_v,
                            __bf16* __restrict__ Tkb, __bf16* __restrict__ Tvt) {
  int i = blockIdx.x * blockDim.x + threadIdx.x;
  if (i >= 1056 * 64) return;
  int p = i >> 6, d = i & 63;
  float vk = (p < 1025) ? tab_k[p * 64 + d] : 0.f;
  float vv = (p < 1025) ? tab_v[p * 64 + d] : 0.f;
  Tkb[p * 64 + d]   = (__bf16)vk;
  Tvt[d * 1056 + p] = (__bf16)vv;
}

// ---------------------------------------------------------------------------
// K1/K3: Y = X @ W^T + bias (M=4096,N=512,K=512), bf16 WMMA, double-buffered
// float4 staging (one b128 global load -> cvt -> one b64 ds store).
// mode 0/1: bf16 head-split [b*8+h][s][d] (Q,K); mode 2: transposed [.][d][s]
// (V); mode 3: f32 row-major (output projection).
// ---------------------------------------------------------------------------
__global__ __launch_bounds__(128)
void gemm_xwt(const float* __restrict__ X, const float* __restrict__ W,
              const float* __restrict__ bias,
              __bf16* __restrict__ dst_bf, float* __restrict__ dst_f, int mode) {
  __shared__ __attribute__((aligned(16))) __bf16 As[2][16 * 40];
  __shared__ __attribute__((aligned(16))) __bf16 Bs[2][64 * 40];
  const int tid = threadIdx.x, wave = tid >> 5, lane = tid & 31;
  const int M0 = blockIdx.x * 16;
  const int N0 = blockIdx.y * 64;

  // exactly one float4 of A and four float4s of B per thread per k-step
  const int ia = tid * 4, ra = ia >> 5, ca = ia & 31;

  auto stage = [&](int buf, int k0) {
    v4f xa = *(const v4f*)&X[(size_t)(M0 + ra) * 512 + k0 + ca];
    v4f xb[4];
#pragma unroll
    for (int t = 0; t < 4; ++t) {
      int i = (tid + t * 128) * 4;
      int r = i >> 5, c = i & 31;
      xb[t] = *(const v4f*)&W[(size_t)(N0 + r) * 512 + k0 + c];
    }
    *(v4bf*)&As[buf][ra * 40 + ca] = cvt4(xa);
#pragma unroll
    for (int t = 0; t < 4; ++t) {
      int i = (tid + t * 128) * 4;
      int r = i >> 5, c = i & 31;
      *(v4bf*)&Bs[buf][r * 40 + c] = cvt4(xb[t]);
    }
  };

  stage(0, 0);
  v8f acc = {};
  for (int kt = 0; kt < 16; ++kt) {
    __syncthreads();
    if (kt + 1 < 16) stage((kt + 1) & 1, (kt + 1) * 32);
    v16bf a = load_frag(As[kt & 1], 40);
    v16bf b = load_frag(Bs[kt & 1] + wave * 16 * 40, 40);
    acc = wmma_bf16(a, b, acc);
  }

  const int col = lane & 15, mbase = (lane >> 4) * 8;
  const int ng = N0 + wave * 16 + col;
  const float bval = bias[ng];
#pragma unroll
  for (int j = 0; j < 8; ++j) {
    int mg = M0 + mbase + j;
    float y = acc[j] + bval;
    if (mode == 3) {
      dst_f[(size_t)mg * 512 + ng] = y;
    } else {
      int bb = mg >> 10, s = mg & 1023;     // mg = b*1024 + s
      int h = ng >> 6, d = ng & 63;         // ng = h*64 + d
      int nhb = bb * 8 + h;                 // head-batch index b*H+h
      if (mode == 2) dst_bf[((size_t)nhb * 64 + d) * 1024 + s] = (__bf16)y;
      else           dst_bf[((size_t)nhb * 1024 + s) * 64 + d] = (__bf16)y;
    }
  }
}

// ---------------------------------------------------------------------------
// K2: attention core. One workgroup = (head-batch n, 16-row q-tile).
// Dynamic LDS 175168 B (< 320KB/WGP):
//   qtabw [16][1056] f32 : Q-tile @ tab_k^T (phase 1-2); aliased later as
//   wbuf  [16][1056] bf16: diagonal-bucketed attn for rel-V GEMM
//   attnf [16][1032] f32 : scores -> exp -> normalized attn
//   attnb [16][1040] bf16: attn in bf16 for attn@V WMMA
//   Vs    [4][2][16][32] bf16: per-wave double-buffered async V staging
// ---------------------------------------------------------------------------
__global__ __launch_bounds__(128)
void attn_kernel(const __bf16* __restrict__ Qh, const __bf16* __restrict__ Kh,
                 const __bf16* __restrict__ Vt, const __bf16* __restrict__ Tkb,
                 const __bf16* __restrict__ Tvt,
                 float* __restrict__ attn_out, float* __restrict__ ctx) {
  const int tid = threadIdx.x, wave = tid >> 5, lane = tid & 31;
  const int Q0 = blockIdx.x * 16;   // q-tile
  const int n  = blockIdx.y;        // head-batch (= b*H + h)

  extern __shared__ unsigned char smem[];
  float*  qtabw = (float*)smem;                            // 67584 B
  __bf16* wbuf  = (__bf16*)smem;                           // alias
  float*  attnf = (float*)(smem + 67584);                  // 66048 B
  __bf16* attnb = (__bf16*)(smem + 67584 + 66048);         // 33280 B
  float*  rinv  = (float*)(smem + 166912);                 // 64 B
  __bf16* Vs    = (__bf16*)(smem + 166976);                // 8192 B

  const int col = lane & 15, mbase = (lane >> 4) * 8;

  // Q tile as two A-fragments (D=64 -> two K=32 chunks); reused everywhere.
  const __bf16* qbase = Qh + ((size_t)n * 1024 + Q0) * 64;
  v16bf a0 = load_frag(qbase, 64);
  v16bf a1 = load_frag(qbase + 32, 64);

  // --- Phase 1: qtab[m][p] = Q[m,:] . tab_k[p,:]  (66 p-tiles over 4 waves)
  for (int pt = wave; pt < 66; pt += 4) {
    int p0 = pt * 16;
    v16bf b0 = load_frag(Tkb + (size_t)p0 * 64, 64);
    v16bf b1 = load_frag(Tkb + (size_t)p0 * 64 + 32, 64);
    v8f acc = {};
    acc = wmma_bf16(a0, b0, acc);
    acc = wmma_bf16(a1, b1, acc);
#pragma unroll
    for (int j = 0; j < 8; ++j) qtabw[(mbase + j) * 1056 + p0 + col] = acc[j];
  }
  __syncthreads();

  // --- Phase 2: scores = (Q K^T + gather(qtab)) / sqrt(D)
  for (int kt = wave; kt < 64; kt += 4) {
    int k0 = kt * 16;
    const __bf16* kb = Kh + ((size_t)n * 1024 + k0) * 64;
    v16bf b0 = load_frag(kb, 64);
    v16bf b1 = load_frag(kb + 32, 64);
    v8f acc = {};
    acc = wmma_bf16(a0, b0, acc);
    acc = wmma_bf16(a1, b1, acc);
    int k = k0 + col;
#pragma unroll
    for (int j = 0; j < 8; ++j) {
      int m = mbase + j;
      int rel = k - (Q0 + m);
      rel = rel < -512 ? -512 : (rel > 512 ? 512 : rel);
      attnf[m * 1032 + k] = (acc[j] + qtabw[m * 1056 + rel + 512]) * 0.125f;
    }
  }
  __syncthreads();

  // --- Phase 3: softmax. 8 lanes/row, lane-interleaved float4 columns
  // (32 consecutive dwords across the 8 lanes -> conflict-free b128 ds loads).
  {
    int r = tid >> 3, l8 = tid & 7;
    float* row = attnf + r * 1032;
    v4f mx4 = {-3.0e38f, -3.0e38f, -3.0e38f, -3.0e38f};
#pragma unroll 4
    for (int j = 0; j < 32; ++j)
      mx4 = vmax4(mx4, *(const v4f*)&row[(j * 8 + l8) * 4]);
    float mx = fmaxf(fmaxf(mx4[0], mx4[1]), fmaxf(mx4[2], mx4[3]));
    mx = fmaxf(mx, __shfl_xor(mx, 1, 32));
    mx = fmaxf(mx, __shfl_xor(mx, 2, 32));
    mx = fmaxf(mx, __shfl_xor(mx, 4, 32));
    v4f s4 = {};
#pragma unroll 4
    for (int j = 0; j < 32; ++j) {
      float* p = &row[(j * 8 + l8) * 4];
      v4f x = *(const v4f*)p, e;
      e[0] = __expf(x[0] - mx); e[1] = __expf(x[1] - mx);
      e[2] = __expf(x[2] - mx); e[3] = __expf(x[3] - mx);
      *(v4f*)p = e;
      s4 += e;
    }
    float sum = (s4[0] + s4[1]) + (s4[2] + s4[3]);
    sum += __shfl_xor(sum, 1, 32);
    sum += __shfl_xor(sum, 2, 32);
    sum += __shfl_xor(sum, 4, 32);
    if ((tid & 7) == 0) rinv[r] = 1.0f / sum;
  }
  __syncthreads();

  // Normalize; single coalesced float4 stream of attn to HBM; keep f32+bf16.
  {
    float* aout = attn_out + ((size_t)n * 1024 + Q0) * 1024;
    for (int i4 = tid * 4; i4 < 16 * 1024; i4 += 512) {
      int r = i4 >> 10, k = i4 & 1023;
      v4f a = *(const v4f*)&attnf[r * 1032 + k] * rinv[r];
      *(v4f*)&attnf[r * 1032 + k] = a;
      *(v4bf*)&attnb[r * 1040 + k] = cvt4(a);
      *(v4f*)&aout[(size_t)r * 1024 + k] = a;
    }
  }
  __syncthreads();

  // --- Phase 4: bucket attn along diagonals: w[m][p] = sum_{k:idx(q,k)=p}
  for (int i = tid; i < 16 * 1056; i += 128) {
    int m = i / 1056, p = i - m * 1056;
    if (p == 0 || p == 1024) continue;   // clip sums written below
    float v = 0.f;
    if (p < 1024) {
      int k = Q0 + m + p - 512;
      if (k >= 0 && k < 1024) v = attnf[m * 1032 + k];
    }
    wbuf[m * 1056 + p] = (__bf16)v;
  }
  {
    int g = tid >> 2, l4 = tid & 3;  // 32 groups of 4 lanes: 16 rows x 2 sides
    int row = g >> 1, side = g & 1;
    float s = 0.f;
    if (side == 0) {
      int kmax = Q0 + row - 512;     // k - q <= -512
      for (int k = l4; k <= kmax; k += 4) s += attnf[row * 1032 + k];
    } else {
      for (int k = Q0 + row + 512 + l4; k < 1024; k += 4)  // k - q >= 512
        s += attnf[row * 1032 + k];
    }
    s += __shfl_xor(s, 1, 32);
    s += __shfl_xor(s, 2, 32);
    if (l4 == 0) wbuf[row * 1056 + (side ? 1024 : 0)] = (__bf16)s;
  }
  __syncthreads();

  // --- Phase 5: out = attn @ V + w @ tab_v  (each wave one 16-wide d-slab)
  {
    int d0 = wave * 16;
    __bf16* Vsw = Vs + wave * 1024;  // 2 x [16][32] per-wave double buffer
    const __bf16* vbase = Vt + ((size_t)n * 64 + d0) * 1024;
    v8f acc = {};
#if ASYNC_CP
    // Pipeline V-slab chunks via ASYNCcnt: copy chunk c+1 while WMMA eats c.
    auto issueV = [&](int buf, int c) {
#pragma unroll
      for (int t = 0; t < 2; ++t) {
        int s = lane + t * 32;       // 64 x 16B segments of the 1KB slab
        int rr = s >> 2, c8 = (s & 3) * 8;
        ASYNC_B128(vbase + (size_t)rr * 1024 + c * 32 + c8,
                   Vsw + buf * 512 + rr * 32 + c8);
      }
    };
    issueV(0, 0);
    for (int c = 0; c < 32; ++c) {
      if (c + 1 < 32) {
        issueV((c + 1) & 1, c + 1);
        WAIT_ASYNC(2);               // in-order: chunk c's 2 copies are done
      } else {
        WAIT_ASYNC(0);
      }
      v16bf aA = load_frag(attnb + c * 32, 1040);
      v16bf bV = load_frag(Vsw + (c & 1) * 512, 32);
      acc = wmma_bf16(aA, bV, acc);
    }
#else
    for (int c = 0; c < 32; ++c) {
      v16bf aA = load_frag(attnb + c * 32, 1040);
      v16bf bV = load_frag(vbase + c * 32, 1024);
      acc = wmma_bf16(aA, bV, acc);
    }
#endif
    for (int c = 0; c < 33; ++c) {   // w @ tab_v (tab_v pre-transposed, L2-hot)
      v16bf aW = load_frag(wbuf + c * 32, 1056);
      v16bf bT = load_frag(Tvt + (size_t)d0 * 1056 + c * 32, 1056);
      acc = wmma_bf16(aW, bT, acc);
    }
    // Reference concat quirk: head-batch n feeds out[b = n%4][s][(n/4)*64+d]
    int b = n & 3, h = n >> 2;
    int d = d0 + col;
    float* cbase = ctx + ((size_t)b * 1024 + Q0) * 512 + h * 64 + d;
#pragma unroll
    for (int j = 0; j < 8; ++j) cbase[(size_t)(mbase + j) * 512] = acc[j];
  }
}

// ---------------------------------------------------------------------------
extern "C" void kernel_launch(void* const* d_in, const int* in_sizes, int n_in,
                              void* d_out, int out_size, void* d_ws, size_t ws_size,
                              hipStream_t stream) {
  (void)in_sizes; (void)n_in; (void)out_size; (void)ws_size;
  const float* query = (const float*)d_in[0];
  const float* key   = (const float*)d_in[1];
  const float* value = (const float*)d_in[2];
  const float* Wq = (const float*)d_in[3];
  const float* bq = (const float*)d_in[4];
  const float* Wk = (const float*)d_in[5];
  const float* bk = (const float*)d_in[6];
  const float* Wv = (const float*)d_in[7];
  const float* bvv = (const float*)d_in[8];
  const float* Wo = (const float*)d_in[9];
  const float* bo = (const float*)d_in[10];
  const float* tab_k = (const float*)d_in[11];
  const float* tab_v = (const float*)d_in[12];

  char* ws = (char*)d_ws;                        // ~21 MB used
  __bf16* Qh  = (__bf16*)(ws + 0);               // [32][1024][64] bf16, 4 MiB
  __bf16* Kh  = (__bf16*)(ws + ((size_t)4 << 20));
  __bf16* Vt  = (__bf16*)(ws + ((size_t)8 << 20));   // [32][64][1024]
  float*  ctx = (float*)(ws + ((size_t)12 << 20));   // [4][1024][512] f32
  __bf16* Tkb = (__bf16*)(ws + ((size_t)20 << 20));            // [1056][64]
  __bf16* Tvt = (__bf16*)(ws + ((size_t)20 << 20) + (256 << 10)); // [64][1056]

  float* out_proj = (float*)d_out;                          // [4][1024][512]
  float* attn_out = (float*)d_out + (size_t)4 * 1024 * 512; // [32][1024][1024]

  prep_tables<<<dim3(264), 256, 0, stream>>>(tab_k, tab_v, Tkb, Tvt);

  dim3 gP(256, 8);  // M-tiles x N-tiles
  gemm_xwt<<<gP, 128, 0, stream>>>(query, Wq, bq, Qh, nullptr, 0);
  gemm_xwt<<<gP, 128, 0, stream>>>(key,   Wk, bk, Kh, nullptr, 1);
  gemm_xwt<<<gP, 128, 0, stream>>>(value, Wv, bvv, Vt, nullptr, 2);

  const size_t smem = 175168;  // 166976 + 8192 async V buffers (<320KB/WGP)
  (void)hipFuncSetAttribute(reinterpret_cast<const void*>(attn_kernel),
                            hipFuncAttributeMaxDynamicSharedMemorySize, (int)smem);
  attn_kernel<<<dim3(64, 32), 128, smem, stream>>>(Qh, Kh, Vt, Tkb, Tvt,
                                                   attn_out, ctx);

  gemm_xwt<<<gP, 128, 0, stream>>>(ctx, Wo, bo, nullptr, out_proj, 3);
}
